// THNN_layer_no_addingones_90185723281665
// MI455X (gfx1250) — matrix operand order repaired and checked by reading, not compiled, not run on hardware
//
#include <hip/hip_runtime.h>
#include <hip/hip_bf16.h>
#include <math.h>

// ---------------------------------------------------------------------------
// THNN hypergraph layer for MI455X (gfx1250, wave32, WMMA, TDM)
//   Prep   : convert f32 weights -> f16 fragments pre-swizzled into the WMMA
//            B-matrix lane layout (one contiguous 32B load per lane per frag)
//   Kernel1: fused node MLP  (scaled + emb2) -- 3 WMMA GEMM phases
//   Kernel2: edge kernel -- Wq frags DMA'd to LDS via tensor_load_to_lds,
//            gather, excl-prod, tanh, WMMA, atomic scatter (segment sum)
//   Kernel0/3: zero accumulator / relu(x/8) finalize
// ---------------------------------------------------------------------------

typedef __attribute__((ext_vector_type(16))) _Float16     v16h;
typedef __attribute__((ext_vector_type(8)))  float        v8f;
typedef __attribute__((ext_vector_type(4)))  unsigned int u32x4;
typedef __attribute__((ext_vector_type(4)))  int          i32x4;
typedef __attribute__((ext_vector_type(8)))  int          i32x8;

#define NN   100000
#define EE   200000
#define KK   4
#define FEAT 256
#define RANK 64
#define HID  512
#define OUTD 128

__constant__ constexpr float NUMC    = 1.6817928305074290861f;  // 8^(1/4)
__constant__ constexpr float INVFACT = 1.0f / 6.0f;             // 1/(K-1)!

// ---- WMMA fragment helpers (16x16x32 f16, wave32) --------------------------
// A (16x32 f16): lane m<16 holds row m, K = {0..7, 16..23}; lane m+16 holds
// row m, K = {8..15, 24..31}
__device__ __forceinline__ v16h load_a_frag(const _Float16* base, int lda, int lane) {
    const int row = lane & 15;
    const int kh  = (lane >> 4) << 3;
    const _Float16* p = base + row * lda + kh;
    v16h a;
#pragma unroll
    for (int i = 0; i < 8; ++i) a[i]     = p[i];
#pragma unroll
    for (int i = 0; i < 8; ++i) a[8 + i] = p[16 + i];
    return a;
}

// Pre-swizzled B fragment: [frag][lane][16 halves] contiguous -> 32B/lane load
__device__ __forceinline__ v16h load_b_frag_sw(const _Float16* fragbase, int lane) {
    return *(const v16h*)(fragbase + lane * 16);
}

// ---------------------------------------------------------------------------
// Prep: swizzle f32 weight [Kd x Nc] into f16 WMMA-B fragments.
// frag (kt, nt) stored at ((nt*KT + kt) * 512) halves; within a fragment,
// lane L half i = W[kt*32 + (L>>4)*8 + i + (i>>3)*8][nt*16 + (L&15)]
// ---------------------------------------------------------------------------
__global__ __launch_bounds__(256) void swizzle_weight_kernel(
    const float* __restrict__ W, _Float16* __restrict__ out, int Kd, int Nc) {
    const int g  = blockIdx.x * 256 + threadIdx.x;
    const int KT = Kd / 32;
    const int nf = KT * (Nc / 16);
    if (g >= nf * 32) return;
    const int frag = g >> 5, lane = g & 31;
    const int nt = frag / KT, kt = frag % KT;
    const int col = nt * 16 + (lane & 15);
    const int kr  = kt * 32 + ((lane >> 4) << 3);
    _Float16* dst = out + (size_t)frag * 512 + lane * 16;
#pragma unroll
    for (int i = 0; i < 16; ++i) dst[i] = (_Float16)W[(size_t)(kr + i + ((i >> 3) << 3)) * Nc + col];
}

// ---------------------------------------------------------------------------
// Kernel 1: fused node MLP.
//   scaled = NUMC * (X @ Wp + bp)            [N x 64]
//   emb2   = relu(X @ W1 + b1) @ W2 + b2     [N x 128]
// ---------------------------------------------------------------------------
__global__ __launch_bounds__(256) void node_mlp_kernel(
    const float* __restrict__ X,
    const _Float16* __restrict__ WpF, const float* __restrict__ bp,
    const _Float16* __restrict__ W1F, const float* __restrict__ b1,
    const _Float16* __restrict__ W2F, const float* __restrict__ b2,
    float* __restrict__ scaled, float* __restrict__ emb2) {

    __shared__ _Float16 Xs[32][FEAT + 8];   // 16.5 KB
    __shared__ _Float16 Hs[32][HID + 8];    // 32.5 KB

    const int tid  = threadIdx.x;
    const int lane = tid & 31;
    const int wave = tid >> 5;
    const int m0   = blockIdx.x * 32;

    // ---- Phase 0: stage X tile, f32 -> f16 --------------------------------
    for (int idx = tid; idx < 32 * (FEAT / 4); idx += 256) {
        const int r  = idx / (FEAT / 4);
        const int c4 = (idx % (FEAT / 4)) * 4;
        const int grow = m0 + r;
        float4 v = make_float4(0.f, 0.f, 0.f, 0.f);
        if (grow < NN) v = *(const float4*)(X + (size_t)grow * FEAT + c4);
        Xs[r][c4 + 0] = (_Float16)v.x;
        Xs[r][c4 + 1] = (_Float16)v.y;
        Xs[r][c4 + 2] = (_Float16)v.z;
        Xs[r][c4 + 3] = (_Float16)v.w;
    }
    __syncthreads();

    const int crow = (lane >> 4) << 3;
    const int ccol = lane & 15;

    // ---- Phase 1: scaled = NUMC*(X @ Wp + bp)   (8 tiles, 1/wave) ---------
    {
        const int mt = wave >> 2, nt = wave & 3;
        v8f acc = {};
#pragma unroll
        for (int k = 0; k < FEAT / 32; ++k) {
            v16h a = load_a_frag(&Xs[mt * 16][k * 32], FEAT + 8, lane);
            v16h b = load_b_frag_sw(WpF + (size_t)(nt * 8 + k) * 512, lane);
            acc = __builtin_amdgcn_wmma_f32_16x16x32_f16(false, a, false, b,
                                                         (short)0, acc, false, false);
        }
        const int col  = nt * 16 + ccol;
        const float bb = bp[col];
#pragma unroll
        for (int i = 0; i < 8; ++i) {
            const int grow = m0 + mt * 16 + crow + i;
            if (grow < NN) scaled[(size_t)grow * RANK + col] = NUMC * (acc[i] + bb);
        }
    }

    // ---- Phase 2: Hs = relu(X @ W1 + b1) f16    (64 tiles, 8/wave) --------
#pragma unroll
    for (int t8 = 0; t8 < 8; ++t8) {
        const int t  = wave + (t8 << 3);
        const int mt = t >> 5, nt = t & 31;
        const _Float16* wslice = W1F + (size_t)nt * 8 * 512;
        __builtin_prefetch(wslice, 0, 1);       // global_prefetch_b8 on W1 slice
        v8f acc = {};
#pragma unroll
        for (int k = 0; k < FEAT / 32; ++k) {
            v16h a = load_a_frag(&Xs[mt * 16][k * 32], FEAT + 8, lane);
            v16h b = load_b_frag_sw(wslice + (size_t)k * 512, lane);
            acc = __builtin_amdgcn_wmma_f32_16x16x32_f16(false, a, false, b,
                                                         (short)0, acc, false, false);
        }
        const int col  = nt * 16 + ccol;
        const float bb = b1[col];
#pragma unroll
        for (int i = 0; i < 8; ++i) {
            const float h = acc[i] + bb;
            Hs[mt * 16 + crow + i][col] = (_Float16)(h > 0.f ? h : 0.f);
        }
    }
    __syncthreads();

    // ---- Phase 3: emb2 = Hs @ W2 + b2           (16 tiles, 2/wave) --------
#pragma unroll
    for (int t2 = 0; t2 < 2; ++t2) {
        const int t  = wave + (t2 << 3);
        const int mt = t >> 3, nt = t & 7;
        const _Float16* wslice = W2F + (size_t)nt * 16 * 512;
        v8f acc = {};
#pragma unroll
        for (int k = 0; k < HID / 32; ++k) {
            v16h a = load_a_frag(&Hs[mt * 16][k * 32], HID + 8, lane);
            v16h b = load_b_frag_sw(wslice + (size_t)k * 512, lane);
            acc = __builtin_amdgcn_wmma_f32_16x16x32_f16(false, a, false, b,
                                                         (short)0, acc, false, false);
        }
        const int col  = nt * 16 + ccol;
        const float bb = b2[col];
#pragma unroll
        for (int i = 0; i < 8; ++i) {
            const int grow = m0 + mt * 16 + crow + i;
            if (grow < NN) emb2[(size_t)grow * OUTD + col] = acc[i] + bb;
        }
    }
}

// ---------------------------------------------------------------------------
// Kernel 2: edge kernel. 16 edges (64 flat member-rows) per block.
// Swizzled Wq fragments (16 KB) are DMA'd into LDS with the Tensor Data
// Mover; all B-fragment reads then hit LDS (ds_load_b128).
// ---------------------------------------------------------------------------
__global__ __launch_bounds__(256) void edge_kernel(
    const int*   __restrict__ edge_nodes,
    const float* __restrict__ scaled, const float* __restrict__ emb2,
    const _Float16* __restrict__ WqF, const float* __restrict__ bq,
    float* __restrict__ out_accum) {

    __shared__ float    Gs[64][RANK];        // 16 KB gathered scaled
    __shared__ _Float16 Ts[64][RANK + 8];    //  9 KB term (f16)
    __shared__ float    E2s[16][OUTD];       //  8 KB relu(edge sum of emb2)
    __shared__ _Float16 WqS[16 * 512];       // 16 KB swizzled Wq fragments
    __shared__ int      Ns[64];

    const int tid  = threadIdx.x;
    const int lane = tid & 31;
    const int wave = tid >> 5;
    const int row0 = blockIdx.x * 64;

#if __has_builtin(__builtin_amdgcn_tensor_load_to_lds)
    if (wave == 0) {
        // ---- Tensor DMA descriptor: 1-D copy of 8192 f16 (16 KB) ----------
        const unsigned long long ga = (unsigned long long)(size_t)WqF;
        const unsigned lds_off = (unsigned)(size_t)(void*)&WqS[0]; // LDS byte offset
        u32x4 g0;
        g0[0] = 1u;                                   // count=1, no gather
        g0[1] = lds_off;                              // lds_addr
        g0[2] = (unsigned)(ga & 0xffffffffu);         // global_addr[31:0]
        g0[3] = (unsigned)((ga >> 32) & 0x01ffffffu)  // global_addr[56:32]
              | 0x80000000u;                          // type=2 (image)
        i32x8 g1;
        g1[0] = 0x00010000;                           // data_size=1 (2 bytes)
        g1[1] = 0x20000000;                           // tensor_dim0 = 8192
        g1[2] = 0x00010000;                           // tensor_dim1 = 1
        g1[3] = 0x20000000;                           // tile_dim0  = 8192
        g1[4] = 0x00000001;                           // tile_dim1 = 1
        g1[5] = 8192;                                 // tensor_dim0_stride
        g1[6] = 0; g1[7] = 0;
        i32x4 z4 = {0, 0, 0, 0};
#if defined(__clang_major__) && (__clang_major__ >= 23)
        i32x8 z8 = {0, 0, 0, 0, 0, 0, 0, 0};
        __builtin_amdgcn_tensor_load_to_lds(g0, g1, z4, z4, z8, 0);
#else
        __builtin_amdgcn_tensor_load_to_lds(g0, g1, z4, z4, 0);
#endif
        __builtin_amdgcn_s_wait_tensorcnt(0);
    }
#else
    for (int idx = tid; idx < 16 * 512 / 8; idx += 256)
        ((float4*)WqS)[idx] = ((const float4*)WqF)[idx];
#endif

    if (tid < 64) Ns[tid] = edge_nodes[row0 + tid];
    __syncthreads();

    // gather scaled rows (coalesced along rank dim)
    for (int idx = tid; idx < 64 * RANK; idx += 256) {
        const int r = idx / RANK, c = idx % RANK;
        Gs[r][c] = scaled[(size_t)Ns[r] * RANK + c];
    }
    // relu of per-edge emb2 sums
    for (int idx = tid; idx < 16 * OUTD; idx += 256) {
        const int e = idx / OUTD, c = idx % OUTD;
        float s = 0.f;
#pragma unroll
        for (int j = 0; j < KK; ++j)
            s += emb2[(size_t)Ns[(e << 2) + j] * OUTD + c];
        E2s[e][c] = s > 0.f ? s : 0.f;
    }
    __syncthreads();

    // term = tanh(INVFACT*NUMC * prod_{j != k} g_j)
    for (int idx = tid; idx < 64 * RANK; idx += 256) {
        const int r = idx / RANK, c = idx % RANK;
        const int e = r >> 2,     k = r & 3;
        float p = 1.f;
#pragma unroll
        for (int j = 0; j < KK; ++j)
            if (j != k) p *= Gs[(e << 2) + j][c];
        Ts[r][c] = (_Float16)tanhf(INVFACT * NUMC * p);
    }
    __syncthreads();

    const int crow = (lane >> 4) << 3;
    const int ccol = lane & 15;

    // qout = term @ Wq ; add bias + relu(edge sum) ; scatter  (32 tiles, 4/wave)
#pragma unroll
    for (int t4 = 0; t4 < 4; ++t4) {
        const int t  = wave + (t4 << 3);
        const int mt = t >> 3, nt = t & 7;
        v8f acc = {};
#pragma unroll
        for (int k = 0; k < RANK / 32; ++k) {
            v16h a = load_a_frag(&Ts[mt * 16][k * 32], RANK + 8, lane);
            v16h b = load_b_frag_sw(&WqS[(nt * 2 + k) * 512], lane);
            acc = __builtin_amdgcn_wmma_f32_16x16x32_f16(false, a, false, b,
                                                         (short)0, acc, false, false);
        }
        const int col  = nt * 16 + ccol;
        const float bb = bq[col];
#pragma unroll
        for (int i = 0; i < 8; ++i) {
            const int r = mt * 16 + crow + i;
            const int e = r >> 2;
            const float v = acc[i] + bb + E2s[e][col];
            atomicAdd(&out_accum[(size_t)Ns[r] * OUTD + col], v);  // global_atomic_add_f32
        }
    }
}

// ---------------------------------------------------------------------------
__global__ __launch_bounds__(256) void zero_kernel(float4* __restrict__ p, int n4) {
    const int i = blockIdx.x * 256 + threadIdx.x;
    if (i < n4) p[i] = make_float4(0.f, 0.f, 0.f, 0.f);
}

__global__ __launch_bounds__(256) void finalize_kernel(float4* __restrict__ p, int n4) {
    const int i = blockIdx.x * 256 + threadIdx.x;
    if (i < n4) {
        float4 v = p[i];
        v.x = fmaxf(v.x * 0.125f, 0.f);
        v.y = fmaxf(v.y * 0.125f, 0.f);
        v.z = fmaxf(v.z * 0.125f, 0.f);
        v.w = fmaxf(v.w * 0.125f, 0.f);
        p[i] = v;
    }
}

// ---------------------------------------------------------------------------
extern "C" void kernel_launch(void* const* d_in, const int* in_sizes, int n_in,
                              void* d_out, int out_size, void* d_ws, size_t ws_size,
                              hipStream_t stream) {
    const float* X  = (const float*)d_in[0];   // embedding [N,256]
    const float* Wp = (const float*)d_in[1];   // [256,64]
    const float* bp = (const float*)d_in[2];   // [64]
    const float* Wq = (const float*)d_in[3];   // [64,128]
    const float* bq = (const float*)d_in[4];   // [128]
    const float* W1 = (const float*)d_in[5];   // [256,512]
    const float* b1 = (const float*)d_in[6];   // [512]
    const float* W2 = (const float*)d_in[7];   // [512,128]
    const float* b2 = (const float*)d_in[8];   // [128]
    const int* edge_nodes = (const int*)d_in[9]; // [E,4] int32

    float* out    = (float*)d_out;                 // [N,128] f32
    float* scaled = (float*)d_ws;                  // [N,64]  f32
    float* emb2   = scaled + (size_t)NN * RANK;    // [N,128] f32
    _Float16* WpF = (_Float16*)(emb2 + (size_t)NN * OUTD); // 16384 halves
    _Float16* W1F = WpF + 256 * 64;                // 131072 halves
    _Float16* W2F = W1F + 256 * 512;               // 65536 halves
    _Float16* WqF = W2F + 512 * 128;               // 8192 halves

    // ---- weight prep: f32 -> f16 pre-swizzled WMMA-B fragments ------------
    swizzle_weight_kernel<<<(32 * 32 + 255) / 256, 256, 0, stream>>>(Wp, WpF, FEAT, RANK);
    swizzle_weight_kernel<<<(256 * 32 + 255) / 256, 256, 0, stream>>>(W1, W1F, FEAT, HID);
    swizzle_weight_kernel<<<(128 * 32 + 255) / 256, 256, 0, stream>>>(W2, W2F, HID, OUTD);
    swizzle_weight_kernel<<<(16 * 32 + 255) / 256, 256, 0, stream>>>(Wq, WqF, RANK, OUTD);

    const int n4 = NN * OUTD / 4;                  // 3,200,000 float4s
    zero_kernel<<<(n4 + 255) / 256, 256, 0, stream>>>((float4*)out, n4);
    node_mlp_kernel<<<(NN + 31) / 32, 256, 0, stream>>>(X, WpF, bp, W1F, b1, W2F, b2,
                                                        scaled, emb2);
    edge_kernel<<<(EE * KK) / 64, 256, 0, stream>>>(edge_nodes, scaled, emb2,
                                                    WqF, bq, out);
    finalize_kernel<<<n4 / 256, 256, 0, stream>>>((float4*)out, n4);
}
